// SchNet_31559419691083
// MI455X (gfx1250) — matrix operand order repaired and checked by reading, not compiled, not run on hardware
//
#include <hip/hip_runtime.h>
#include <hip/hip_bf16.h>

// ---------------- constants (match reference) ----------------
#define Hc   128
#define Rr   50
#define NBc  2
#define NGc  64
#define CUTc 5.0f
#define PIf  3.14159265358979323846f

typedef __attribute__((ext_vector_type(16))) _Float16 v16h;
typedef __attribute__((ext_vector_type(8)))  float    v8f;

// ---------------- fast math (gfx1250 TRANS ops) ----------------
__device__ __forceinline__ float fast_tanh(float x) {
#if __has_builtin(__builtin_amdgcn_tanhf)
  return __builtin_amdgcn_tanhf(x);
#elif __has_builtin(__builtin_amdgcn_tanh_f32)
  return __builtin_amdgcn_tanh_f32(x);
#else
  return tanhf(x);
#endif
}

// ---------------- WMMA helpers (gfx1250 wave32) ----------------
// D = A(16x32 f16) * B(32x16 f16) + C(16x16 f32)
__device__ __forceinline__ v8f wmma_f16(v16h a, v16h b, v8f c) {
  return __builtin_amdgcn_wmma_f32_16x16x32_f16(
      /*neg_a=*/false, a, /*neg_b=*/false, b,
      /*c_mod=*/(short)0, c, /*reuse_a=*/false, /*reuse_b=*/false);
}

// A-fragment (16x32, f16) from row-major [M x lda] at (m0,k0).
// ISA layout: lane L holds row M=L%16; elements 0..7 -> K = kb+0..7,
// elements 8..15 -> K = kb+16..23, kb = k0 + 8*(L>=16).
__device__ __forceinline__ v16h loadA_f16(const _Float16* A, int lda, int m0, int k0) {
  const int lane = threadIdx.x & 31;
  const _Float16* p = A + (long)(m0 + (lane & 15)) * lda + k0 + ((lane >> 4) << 3);
  v16h a;
#pragma unroll
  for (int e = 0; e < 8; ++e) { a[e] = p[e]; a[8 + e] = p[16 + e]; }
  return a;
}
__device__ __forceinline__ v16h loadA_f32(const float* A, int lda, int m0, int k0) {
  const int lane = threadIdx.x & 31;
  const float* p = A + (long)(m0 + (lane & 15)) * lda + k0 + ((lane >> 4) << 3);
  v16h a;
#pragma unroll
  for (int e = 0; e < 8; ++e) { a[e] = (_Float16)p[e]; a[8 + e] = (_Float16)p[16 + e]; }
  return a;
}

// Pre-swizzled B fragments: for weight W[K x 128], fragment (kt, n) holds the
// 32x16 tile at (kt*32, n*16) laid out [lane][16 halves] contiguously, so a
// wave loads its whole fragment as one aligned 32B-per-lane vector load.
__device__ __forceinline__ v16h loadB_sw(const _Float16* Bsw, int kt, int n) {
  const int lane = threadIdx.x & 31;
  return *(const v16h*)(Bsw + ((((long)kt * 8 + n) * 32 + lane) << 4));
}
// C/D f32 16x16: lane L -> col N=L%16, element r -> row M = r + 8*(L>=16).

// ---------------- small utility kernels ----------------
__global__ void k_zero(float* p, int n) {
  for (int i = blockIdx.x * blockDim.x + threadIdx.x; i < n; i += gridDim.x * blockDim.x)
    p[i] = 0.0f;
}

// Swizzle one f32 weight [K x 128] into WMMA B-fragment order (f16).
// kpad: logical K (multiple of 32); ksrc: valid rows in source (rest -> 0).
__global__ void k_swzB(const float* __restrict__ W, _Float16* __restrict__ out,
                       int kpad, int ksrc) {
  const int total = (kpad / 32) * 8 * 32 * 16;
  for (int i = blockIdx.x * blockDim.x + threadIdx.x; i < total;
       i += gridDim.x * blockDim.x) {
    const int frag = i >> 9;          // /512
    const int rem  = i & 511;
    const int lane = rem >> 4;
    const int e    = rem & 15;
    const int kt   = frag >> 3;
    const int n    = frag & 7;
    const int k    = kt * 32 + (e & 7) + ((lane >= 16) ? 8 : 0) + ((e >= 8) ? 16 : 0);
    const int col  = n * 16 + (lane & 15);
    out[i] = (_Float16)((k < ksrc) ? W[(long)k * Hc + col] : 0.0f);
  }
}

__global__ void k_embed(const int* __restrict__ types, const float* __restrict__ emb,
                        float* __restrict__ x, int N) {
  int i = blockIdx.x * blockDim.x + threadIdx.x;
  if (i >= N * Hc) return;
  int n = i >> 7, h = i & (Hc - 1);
  x[i] = emb[(long)types[n] * Hc + h];
}

// ---------------- node GEMM (compile-time MODE, no runtime branches) ----------------
// MODE 0: Ch = A_f32 @ W                      (f16 out, no bias)
// MODE 1: Ch = tanh(A_f32 @ W + bias)         (f16 out)
// MODE 2: xio = xio + (A_f16 @ W + bias)      (f32 in-place residual)
// One wave computes a 16x128 row stripe (A fragments reused across 8 N-tiles).
template <int MODE>
__global__ __launch_bounds__(128) void k_ngemm(
    const void* __restrict__ Ain, const _Float16* __restrict__ Wsw,
    const float* __restrict__ bias, float* __restrict__ xio,
    _Float16* __restrict__ Ch, int M)
{
  const int lane = threadIdx.x & 31;
  const int wv   = threadIdx.x >> 5;
  const int m0   = (blockIdx.x * 4 + wv) * 16;
  if (m0 >= M) return;                      // wave-uniform (M % 16 == 0)

  v16h a[4];
  if (MODE == 2) {
    const _Float16* A = (const _Float16*)Ain;
#pragma unroll
    for (int k = 0; k < 4; ++k) a[k] = loadA_f16(A, Hc, m0, k * 32);
  } else {
    const float* A = (const float*)Ain;
#pragma unroll
    for (int k = 0; k < 4; ++k) a[k] = loadA_f32(A, Hc, m0, k * 32);
  }
#pragma unroll
  for (int n = 0; n < 8; ++n) {
    v16h b[4];
#pragma unroll
    for (int k = 0; k < 4; ++k) b[k] = loadB_sw(Wsw, k, n);   // 8 b128 in one clause
    v8f acc = {};
#pragma unroll
    for (int k = 0; k < 4; ++k) acc = wmma_f16(a[k], b[k], acc);
    const int hc = n * 16 + (lane & 15);
    const float bb = (MODE == 0) ? 0.0f : bias[hc];
#pragma unroll
    for (int r = 0; r < 8; ++r) {
      const long row = m0 + r + ((lane >> 4) << 3);
      float v = acc[r] + bb;
      if (MODE == 1) v = fast_tanh(v);
      if (MODE == 2) {
        xio[row * Hc + hc] += v;
      } else {
        Ch[row * Hc + hc] = (_Float16)v;
      }
    }
  }
}

// ---------------- fused edge kernel ----------------
// Per wave: 16 edges. dist -> RBF(LDS,f16, K padded 64) -> WMMA MLP (tanh between)
// -> * cosine cutoff -> * gathered h16[src] (staged in LDS) -> atomicAdd agg[dst].
__global__ __launch_bounds__(128) void k_edge(
    const float* __restrict__ pos, const int* __restrict__ src, const int* __restrict__ dst,
    const _Float16* __restrict__ fw1sw, const float* __restrict__ fb1,
    const _Float16* __restrict__ fw2sw, const float* __restrict__ fb2,
    const _Float16* __restrict__ h16, float* __restrict__ agg, int E)
{
  __shared__ _Float16 s_rbf[4][16][64];     //  8 KB
  __shared__ _Float16 s_T[4][16][Hc];       // 16 KB (filter tmp, then gathered h)
  __shared__ float    s_d[4][16];
  __shared__ float    s_cv[4][16];
  __shared__ int      s_src[4][16];
  __shared__ int      s_dst[4][16];

  const int lane = threadIdx.x & 31;
  const int wv   = threadIdx.x >> 5;
  const int e0   = (blockIdx.x * 4 + wv) * 16;
  if (e0 >= E) return;                      // wave-uniform (E % 16 == 0)

  if (lane < 16) {
    const int e = e0 + lane;
    const int s = src[e], t = dst[e];
    s_src[wv][lane] = s; s_dst[wv][lane] = t;
    const float dx = pos[3 * s]     - pos[3 * t];
    const float dy = pos[3 * s + 1] - pos[3 * t + 1];
    const float dz = pos[3 * s + 2] - pos[3 * t + 2];
    const float d = sqrtf(dx * dx + dy * dy + dz * dz + 1e-12f);
    s_d[wv][lane]  = d;
    s_cv[wv][lane] = (d < CUTc) ? 0.5f * (__cosf(d * (PIf / CUTc)) + 1.0f) : 0.0f;
  }
  __syncthreads();

  // Gaussian RBF expansion into LDS (f16), K=50 padded to 64 with zeros.
  const float delta = CUTc / (float)(Rr - 1);
  const float coeff = -0.5f / (delta * delta);
  for (int t = lane; t < 16 * 64; t += 32) {
    const int e = t >> 6, k = t & 63;
    float v = 0.0f;
    if (k < Rr) { const float u = s_d[wv][e] - (float)k * delta; v = __expf(coeff * u * u); }
    s_rbf[wv][e][k] = (_Float16)v;
  }
  __syncthreads();

  // GEMM1: T = tanh(rbf @ fw1 + fb1)   [16 x 128]
  const v16h a0 = loadA_f16(&s_rbf[wv][0][0], 64, 0, 0);
  const v16h a1 = loadA_f16(&s_rbf[wv][0][0], 64, 0, 32);
#pragma unroll
  for (int n = 0; n < 8; ++n) {
    v16h b0 = loadB_sw(fw1sw, 0, n);
    v16h b1 = loadB_sw(fw1sw, 1, n);
    v8f acc = {};
    acc = wmma_f16(a0, b0, acc);
    acc = wmma_f16(a1, b1, acc);
    const int hc = n * 16 + (lane & 15);
    const float bb = fb1[hc];
#pragma unroll
    for (int r = 0; r < 8; ++r) {
      const int row = r + ((lane >> 4) << 3);
      s_T[wv][row][hc] = (_Float16)fast_tanh(acc[r] + bb);
    }
  }
  __syncthreads();

  // Pull the whole filter tile into registers, then reuse s_T to stage h[src].
  v16h ta[4];
#pragma unroll
  for (int k = 0; k < 4; ++k) ta[k] = loadA_f16(&s_T[wv][0][0], Hc, 0, k * 32);
  __syncthreads();

  // Stage gathered h rows: 16 rows x 256 B, 32 B per lane per step (coalesced).
#pragma unroll
  for (int it = 0; it < 4; ++it) {
    const int row = it * 4 + (lane >> 3);
    const int ch  = (lane & 7) * 16;
    *(v16h*)&s_T[wv][row][ch] =
        *(const v16h*)&h16[(long)s_src[wv][row] * Hc + ch];
  }
  __syncthreads();

  // GEMM2: W = (T @ fw2 + fb2) * Cenv ; msg = W * h[src] ; scatter-add to agg[dst]
#pragma unroll
  for (int n = 0; n < 8; ++n) {
    v16h b[4];
#pragma unroll
    for (int k = 0; k < 4; ++k) b[k] = loadB_sw(fw2sw, k, n);
    v8f acc = {};
#pragma unroll
    for (int k = 0; k < 4; ++k) acc = wmma_f16(ta[k], b[k], acc);
    const int hc = n * 16 + (lane & 15);
    const float bb = fb2[hc];
#pragma unroll
    for (int r = 0; r < 8; ++r) {
      const int row = r + ((lane >> 4) << 3);
      const float Wv  = (acc[r] + bb) * s_cv[wv][row];
      const float msg = Wv * (float)s_T[wv][row][hc];
      atomicAdd(&agg[(long)s_dst[wv][row] * Hc + hc], msg);
    }
  }
}

// ---------------- output head: e = tanh(x@w1+b1)@w2 + b2 ; energy[batch] += e ----------------
__global__ __launch_bounds__(64) void k_out(
    const float* __restrict__ x, const float* __restrict__ w1, const float* __restrict__ b1,
    const float* __restrict__ w2, const float* __restrict__ b2,
    const int* __restrict__ batch, float* __restrict__ energy)
{
  __shared__ float red[64];
  const int n = blockIdx.x, t = threadIdx.x;
  const float* xr = x + (long)n * Hc;
  float s = b1[t];
#pragma unroll 8
  for (int k = 0; k < Hc; ++k) s += xr[k] * w1[k * 64 + t];
  red[t] = fast_tanh(s) * w2[t];
  __syncthreads();
  for (int off = 32; off > 0; off >>= 1) {
    if (t < off) red[t] += red[t + off];
    __syncthreads();
  }
  if (t == 0) atomicAdd(&energy[batch[n]], red[0] + b2[0]);
}

// ---------------- host launcher ----------------
extern "C" void kernel_launch(void* const* d_in, const int* in_sizes, int n_in,
                              void* d_out, int out_size, void* d_ws, size_t ws_size,
                              hipStream_t stream) {
  const int*   atom_types = (const int*)  d_in[0];
  const float* pos        = (const float*)d_in[1];
  const int*   eidx       = (const int*)  d_in[2];
  const int*   batch      = (const int*)  d_in[3];
  const float* emb        = (const float*)d_in[4];
  const float* fw1        = (const float*)d_in[5];
  const float* fb1        = (const float*)d_in[6];
  const float* fw2        = (const float*)d_in[7];
  const float* fb2        = (const float*)d_in[8];
  const float* lin1_w     = (const float*)d_in[9];
  const float* lin2_w     = (const float*)d_in[10];
  const float* lin2_b     = (const float*)d_in[11];
  const float* blk_w      = (const float*)d_in[12];
  const float* blk_b      = (const float*)d_in[13];
  const float* out_w1     = (const float*)d_in[14];
  const float* out_b1     = (const float*)d_in[15];
  const float* out_w2     = (const float*)d_in[16];
  const float* out_b2     = (const float*)d_in[17];

  const int N = in_sizes[0];
  const int E = in_sizes[2] / 2;
  const int* src = eidx;
  const int* dst = eidx + E;

  // workspace carve-up (256B aligned)
  char* ws = (char*)d_ws;
  size_t off = 0;
  auto take = [&](size_t bytes) -> char* {
    char* p = ws + off;
    off = (off + bytes + 255) & ~(size_t)255;
    return p;
  };
  const size_t WSZ  = (size_t)Hc * Hc;       // elements per 128x128 weight
  const size_t W1SZ = (size_t)64 * Hc;       // padded fw1 per block
  float*    x     = (float*)   take((size_t)N * Hc * 4);
  float*    agg   = (float*)   take((size_t)N * Hc * 4);
  _Float16* h16   = (_Float16*)take((size_t)N * Hc * 2);   // h, then reused as t
  _Float16* fw1sw = (_Float16*)take(NBc * W1SZ * 2);
  _Float16* fw2sw = (_Float16*)take(NBc * WSZ * 2);
  _Float16* l1sw  = (_Float16*)take(NBc * WSZ * 2);
  _Float16* l2sw  = (_Float16*)take(NBc * WSZ * 2);
  _Float16* blksw = (_Float16*)take(NBc * WSZ * 2);
  (void)ws_size; (void)n_in; (void)out_size;

  // weight prep: f32 -> f16 swizzled into WMMA B-fragment order
  for (int i = 0; i < NBc; ++i) {
    k_swzB<<<32,  256, 0, stream>>>(fw1    + (size_t)i * Rr * Hc, fw1sw + i * W1SZ, 64, Rr);
    k_swzB<<<64,  256, 0, stream>>>(fw2    + i * WSZ,             fw2sw + i * WSZ, Hc, Hc);
    k_swzB<<<64,  256, 0, stream>>>(lin1_w + i * WSZ,             l1sw  + i * WSZ, Hc, Hc);
    k_swzB<<<64,  256, 0, stream>>>(lin2_w + i * WSZ,             l2sw  + i * WSZ, Hc, Hc);
    k_swzB<<<64,  256, 0, stream>>>(blk_w  + i * WSZ,             blksw + i * WSZ, Hc, Hc);
  }

  // x = emb[atom_types]
  k_embed<<<((N * Hc) + 255) / 256, 256, 0, stream>>>(atom_types, emb, x, N);

  const int gN = (N + 63) / 64;     // node-gemm blocks (4 waves x 16 rows)
  const int gE = (E + 63) / 64;     // edge blocks      (4 waves x 16 edges)

  for (int i = 0; i < NBc; ++i) {
    k_zero<<<1024, 256, 0, stream>>>(agg, N * Hc);
    // h = x @ lin1   (f16 out for the edge gather)
    k_ngemm<0><<<gN, 128, 0, stream>>>(x, l1sw + i * WSZ, nullptr, nullptr, h16, N);
    // fused message pass
    k_edge<<<gE, 128, 0, stream>>>(pos, src, dst,
                                   fw1sw + i * W1SZ, fb1 + (size_t)i * Hc,
                                   fw2sw + i * WSZ,  fb2 + (size_t)i * Hc,
                                   h16, agg, E);
    // t = tanh(agg @ lin2 + b)   (reuse h16 buffer)
    k_ngemm<1><<<gN, 128, 0, stream>>>(agg, l2sw + i * WSZ,
                                       lin2_b + (size_t)i * Hc, nullptr, h16, N);
    // x = x + (t @ blk + b)
    k_ngemm<2><<<gN, 128, 0, stream>>>(h16, blksw + i * WSZ,
                                       blk_b + (size_t)i * Hc, x, nullptr, N);
  }

  // output head + per-graph energy
  k_zero<<<1, 64, 0, stream>>>((float*)d_out, NGc);
  k_out<<<N, 64, 0, stream>>>(x, out_w1, out_b1, out_w2, out_b2, batch, (float*)d_out);
}
